// Autoencoder3_26010321944632
// MI455X (gfx1250) — matrix operand order repaired and checked by reading, not compiled, not run on hardware
//
#include <hip/hip_runtime.h>

typedef __attribute__((ext_vector_type(16))) __bf16 v16bf;
typedef __attribute__((ext_vector_type(8)))  float  v8f;

union BFrag { unsigned u[8]; v16bf v; };

static __device__ __forceinline__ unsigned pack2(__bf16 a, __bf16 b) {
  union { __bf16 h[2]; unsigned u; } x; x.h[0] = a; x.h[1] = b; return x.u;
}
static __device__ __forceinline__ float bflo(unsigned u) {
  union { unsigned short s; __bf16 h; } x; x.s = (unsigned short)(u & 0xffff); return (float)x.h;
}
static __device__ __forceinline__ float bfhi(unsigned u) {
  union { unsigned short s; __bf16 h; } x; x.s = (unsigned short)(u >> 16); return (float)x.h;
}
static __device__ __forceinline__ int iclamp(int v, int lo, int hi) {
  return v < lo ? lo : (v > hi ? hi : v);
}

// tap -> linear offset helpers (fold to constants after unrolling)
#define TOFF18(t) (((t) / 9) * 324 + (((t) / 3) % 3) * 18 + ((t) % 3))
#define TOFF10(t) (((t) / 9) * 100 + (((t) / 3) % 3) * 10 + ((t) % 3))

// ---------------- workspace layout (bytes) ----------------
#define OFF_E1P 0u        // enc1 A-pack  [16][64]  bf16  (K=tap*2+ic, zero pad >=54)
#define OFF_E2P 2048u     // enc2 pack    [32][27][16] bf16
#define OFF_E3P 29696u    // enc3 pack    [64][27][32] bf16
#define OFF_D1P 140288u   // dec1         [32][64] bf16
#define OFF_D2P 144384u   // dec2         [16][32] bf16
#define OFF_F3  145408u   // f3 [3][3][3][64] f32
#define OFF_F2  152320u   // f2 [12][12][12][32] bf16
#define OFF_F1  262912u   // f1 [48][48][48][16] bf16

// ---------------- kernel 0: weight prepack (f32 -> bf16, WMMA-friendly K order) -----
__global__ void k_pack(const float* __restrict__ e1w, const float* __restrict__ e2w,
                       const float* __restrict__ e3w, const float* __restrict__ d1w,
                       const float* __restrict__ d2w,
                       __bf16* __restrict__ e1p, __bf16* __restrict__ e2p,
                       __bf16* __restrict__ e3p, __bf16* __restrict__ d1p,
                       __bf16* __restrict__ d2p) {
  int i = blockIdx.x * blockDim.x + threadIdx.x;
  if (i < 1024) {                       // enc1: [o][K], K = tap*2+ic (54 valid of 64)
    int o = i >> 6, K = i & 63;
    float v = 0.f;
    if (K < 54) { int tap = K >> 1, ic = K & 1; v = e1w[(o * 2 + ic) * 27 + tap]; }
    e1p[i] = (__bf16)v;
    return;
  }
  i -= 1024;
  if (i < 13824) {                      // enc2: [o][tap][ic16]
    int ic = i & 15, tap = (i >> 4) % 27, o = i / (16 * 27);
    e2p[i] = (__bf16)e2w[(o * 16 + ic) * 27 + tap];
    return;
  }
  i -= 13824;
  if (i < 55296) {                      // enc3: [o][tap][ic32]
    int ic = i & 31, tap = (i >> 5) % 27, o = i / (32 * 27);
    e3p[i] = (__bf16)e3w[(o * 32 + ic) * 27 + tap];
    return;
  }
  i -= 55296;
  if (i < 2048) { d1p[i] = (__bf16)d1w[i]; return; }
  i -= 2048;
  if (i < 512)  { d2p[i] = (__bf16)d2w[i]; return; }
}

// ---------------- kernel 1: conv1(2->16) + relu + maxpool4 -> f1[48^3][16] ----------
__global__ void __launch_bounds__(256) k_conv1(const float* __restrict__ x0,
                                               const float* __restrict__ x1,
                                               const __bf16* __restrict__ wA,
                                               const float* __restrict__ bias1,
                                               __bf16* __restrict__ f1) {
  __shared__ __align__(16) unsigned sIn[18 * 18 * 18];  // 1 dword = 2 bf16 channels
  int b = blockIdx.x;
  int bx = b % 12, by = (b / 12) % 12, bz = b / 144;
  int ox = bx * 16, oy = by * 16, oz = bz * 16;
  for (int col = threadIdx.x; col < 324; col += blockDim.x) {
    int lx = col % 18, ly = col / 18;
    int gx = ox + lx - 1, gy = oy + ly - 1;
    bool xyok = ((unsigned)gx < 192u) && ((unsigned)gy < 192u);
    size_t gxy = (size_t)gy * 192 + gx;
    for (int lz = 0; lz < 18; ++lz) {
      int gz = oz + lz - 1;
      unsigned val = 0u;
      if (xyok && (unsigned)gz < 192u) {
        size_t gi = (size_t)gz * 36864 + gxy;
        val = pack2((__bf16)x0[gi], (__bf16)x1[gi]);
      }
      sIn[lz * 324 + col] = val;
    }
  }
  __syncthreads();

  int lane = threadIdx.x & 31, wave = threadIdx.x >> 5;
  int hf = lane >> 4, nn = lane & 15;

  // A fragments (weights), per-lane, loop-invariant
  BFrag A[2];
#pragma unroll
  for (int s = 0; s < 2; ++s)
#pragma unroll
    for (int v = 0; v < 8; ++v) {
      int K = 32 * s + 16 * hf + 2 * v;
      A[s].u[v] = *(const unsigned*)(wA + nn * 64 + K);
    }
  float bias[8];
#pragma unroll
  for (int r = 0; r < 8; ++r) bias[r] = bias1[r + 8 * hf];

  // hoisted per-lane B tap offsets (constant across cells/slices)
  int boff[2][8];
#pragma unroll
  for (int s = 0; s < 2; ++s)
#pragma unroll
    for (int v = 0; v < 8; ++v) {
      int t0 = 16 * s + v;                                   // hf==0 tap (always <27)
      int t1 = (16 * s + 8 + v) < 27 ? (16 * s + 8 + v) : 26;  // hf==1 tap (clamped)
      boff[s][v] = hf ? TOFF18(t1) : TOFF18(t0);
    }

  for (int j = 0; j < 8; ++j) {
    int c = wave * 8 + j;
    int cx = c & 3, cy = (c >> 2) & 3, cz = c >> 4;
    int bxy = (cy * 4 + (nn >> 2)) * 18 + cx * 4 + (nn & 3);
    v8f pool;
#pragma unroll
    for (int r = 0; r < 8; ++r) pool[r] = 0.f;
    for (int zz = 0; zz < 4; ++zz) {
      int base = (cz * 4 + zz) * 324 + bxy;
      v8f acc;
#pragma unroll
      for (int r = 0; r < 8; ++r) acc[r] = bias[r];
#pragma unroll
      for (int s = 0; s < 2; ++s) {
        BFrag B;
#pragma unroll
        for (int v = 0; v < 8; ++v) {
          unsigned val = sIn[base + boff[s][v]];
          if (16 * s + 8 + v > 26) val = hf ? 0u : val;  // compile-time selected lanes only
          B.u[v] = val;
        }
        acc = __builtin_amdgcn_wmma_f32_16x16x32_bf16(false, A[s].v, false, B.v,
                                                      (short)0, acc, false, false);
      }
#pragma unroll
      for (int r = 0; r < 8; ++r) {
        float x = acc[r];
        pool[r] = fmaxf(pool[r], x > 0.f ? x : 0.f);
      }
    }
#pragma unroll
    for (int m = 1; m < 16; m <<= 1)
#pragma unroll
      for (int r = 0; r < 8; ++r) pool[r] = fmaxf(pool[r], __shfl_xor(pool[r], m, 32));
    if (nn == 0) {
      int px = bx * 4 + cx, py = by * 4 + cy, pz = bz * 4 + cz;
      size_t base2 = (((size_t)pz * 48 + py) * 48 + px) * 16;
#pragma unroll
      for (int r = 0; r < 8; ++r) f1[base2 + r + 8 * hf] = (__bf16)pool[r];
    }
  }
}

// ---------------- kernel 2: conv2(16->32) + relu + maxpool4 -> f2[12^3][32] ---------
__global__ void __launch_bounds__(256) k_conv2(const __bf16* __restrict__ f1,
                                               const __bf16* __restrict__ wP,
                                               const float* __restrict__ bias2,
                                               __bf16* __restrict__ f2) {
  __shared__ __align__(16) unsigned sIn[10 * 10 * 10 * 8];  // voxel-major, 8 dword = 16 ch
  __shared__ __align__(16) unsigned sW[32 * 27 * 8];
  int b = blockIdx.x;
  int btx = b % 6, bty = (b / 6) % 6, btz = b / 36;
  int ox = btx * 8, oy = bty * 8, oz = btz * 8;
  const unsigned* f1u = (const unsigned*)f1;
  for (int i = threadIdx.x; i < 1000; i += blockDim.x) {
    int lx = i % 10, ly = (i / 10) % 10, lz = i / 100;
    int gx = ox + lx - 1, gy = oy + ly - 1, gz = oz + lz - 1;
    uint4 lo = {0, 0, 0, 0}, hi = {0, 0, 0, 0};
    if ((unsigned)gx < 48u && (unsigned)gy < 48u && (unsigned)gz < 48u) {
      const uint4* p = (const uint4*)(f1u + (((size_t)gz * 48 + gy) * 48 + gx) * 8);
      lo = p[0]; hi = p[1];
    }
    uint4* q = (uint4*)(sIn + i * 8);
    q[0] = lo; q[1] = hi;
  }
  const unsigned* wPu = (const unsigned*)wP;
  for (int i = threadIdx.x; i < 32 * 27 * 8; i += blockDim.x) sW[i] = wPu[i];
  __syncthreads();

  int lane = threadIdx.x & 31, wave = threadIdx.x >> 5;
  int hf = lane >> 4, nn = lane & 15;
  int cx = wave & 1, cy = (wave >> 1) & 1, cz = (wave >> 2) & 1;

  float bias[2][8];
#pragma unroll
  for (int mt = 0; mt < 2; ++mt)
#pragma unroll
    for (int r = 0; r < 8; ++r) bias[mt][r] = bias2[mt * 16 + r + 8 * hf];

  int aB[2];
#pragma unroll
  for (int mt = 0; mt < 2; ++mt) aB[mt] = (mt * 16 + nn) * 216;

  int bOff[14];
#pragma unroll
  for (int s = 0; s < 14; ++s) {
    int t0 = 2 * s;
    int t1 = (2 * s + 1) < 27 ? (2 * s + 1) : 26;
    bOff[s] = (hf ? TOFF10(t1) : TOFF10(t0)) * 8;
  }

  v8f pool[2];
#pragma unroll
  for (int mt = 0; mt < 2; ++mt)
#pragma unroll
    for (int r = 0; r < 8; ++r) pool[mt][r] = 0.f;

  int bxy = ((cy * 4 + (nn >> 2)) * 10 + cx * 4 + (nn & 3)) * 8;
  for (int zz = 0; zz < 4; ++zz) {
    int lz = cz * 4 + zz;
    int voxbase = lz * 800 + bxy;
    v8f acc[2];
#pragma unroll
    for (int mt = 0; mt < 2; ++mt)
#pragma unroll
      for (int r = 0; r < 8; ++r) acc[mt][r] = bias[mt][r];
#pragma unroll
    for (int s = 0; s < 14; ++s) {
      BFrag B;
      const uint4* p = (const uint4*)(sIn + voxbase + bOff[s]);
      ((uint4*)B.u)[0] = p[0];
      ((uint4*)B.u)[1] = p[1];
      if (s == 13) {  // hf==1 tap would be 27 (invalid)
#pragma unroll
        for (int v = 0; v < 8; ++v) B.u[v] = hf ? 0u : B.u[v];
      }
#pragma unroll
      for (int mt = 0; mt < 2; ++mt) {
        BFrag A;
#pragma unroll
        for (int v = 0; v < 8; ++v) {
          int tap = 2 * s + hf;  // per-half tap
          A.u[v] = (tap < 27) ? sW[aB[mt] + tap * 8 + v] : 0u;
        }
        acc[mt] = __builtin_amdgcn_wmma_f32_16x16x32_bf16(false, A.v, false, B.v,
                                                          (short)0, acc[mt], false, false);
      }
    }
#pragma unroll
    for (int mt = 0; mt < 2; ++mt)
#pragma unroll
      for (int r = 0; r < 8; ++r) {
        float x = acc[mt][r];
        pool[mt][r] = fmaxf(pool[mt][r], x > 0.f ? x : 0.f);
      }
  }
#pragma unroll
  for (int m = 1; m < 16; m <<= 1)
#pragma unroll
    for (int mt = 0; mt < 2; ++mt)
#pragma unroll
      for (int r = 0; r < 8; ++r)
        pool[mt][r] = fmaxf(pool[mt][r], __shfl_xor(pool[mt][r], m, 32));
  if (nn == 0) {
    int px = btx * 2 + cx, py = bty * 2 + cy, pz = btz * 2 + cz;
    size_t base = (((size_t)pz * 12 + py) * 12 + px) * 32;
#pragma unroll
    for (int mt = 0; mt < 2; ++mt)
#pragma unroll
      for (int r = 0; r < 8; ++r) f2[base + mt * 16 + r + 8 * hf] = (__bf16)pool[mt][r];
  }
}

// ---------------- kernel 3: conv3(32->64) + relu + maxpool4 -> f3[3^3][64] f32 ------
__global__ void __launch_bounds__(128) k_conv3(const __bf16* __restrict__ f2,
                                               const __bf16* __restrict__ wP,
                                               const float* __restrict__ bias3,
                                               float* __restrict__ f3) {
  int b = blockIdx.x;  // 0..26 pooled cells
  int px = b % 3, py = (b / 3) % 3, pz = b / 9;
  int lane = threadIdx.x & 31, wave = threadIdx.x >> 5;  // mt = wave (4 M-tiles)
  int hf = lane >> 4, nn = lane & 15;
  int mt = wave;
  const unsigned* wPu = (const unsigned*)wP;
  const unsigned* f2u = (const unsigned*)f2;

  float bias[8];
#pragma unroll
  for (int r = 0; r < 8; ++r) bias[r] = bias3[mt * 16 + r + 8 * hf];
  int aBase = (mt * 16 + nn) * 432 + 8 * hf;
  int lx = px * 4 + (nn & 3), ly = py * 4 + (nn >> 2);
  v8f pool;
#pragma unroll
  for (int r = 0; r < 8; ++r) pool[r] = 0.f;

  for (int zz = 0; zz < 4; ++zz) {
    int lz = pz * 4 + zz;
    v8f acc;
#pragma unroll
    for (int r = 0; r < 8; ++r) acc[r] = bias[r];
#pragma unroll
    for (int s = 0; s < 27; ++s) {  // K = s*32 + ic
      int dz = s / 9, dy = (s / 3) % 3, dx = s % 3;  // constants after unroll
      int gx = lx + dx - 1, gy = ly + dy - 1, gz = lz + dz - 1;
      BFrag A;
#pragma unroll
      for (int v = 0; v < 8; ++v) {
        int cst = v;
        A.u[v] = wPu[aBase + s * 16 + cst];
      }
      BFrag B;
      bool ok = ((unsigned)gx < 12u) && ((unsigned)gy < 12u) && ((unsigned)gz < 12u);
      if (ok) {
        const uint4* p = (const uint4*)(f2u + (((size_t)gz * 12 + gy) * 12 + gx) * 16 + hf * 8);
        ((uint4*)B.u)[0] = p[0];
        ((uint4*)B.u)[1] = p[1];
      } else {
#pragma unroll
        for (int v = 0; v < 8; ++v) B.u[v] = 0u;
      }
      acc = __builtin_amdgcn_wmma_f32_16x16x32_bf16(false, A.v, false, B.v,
                                                    (short)0, acc, false, false);
    }
#pragma unroll
    for (int r = 0; r < 8; ++r) {
      float x = acc[r];
      pool[r] = fmaxf(pool[r], x > 0.f ? x : 0.f);
    }
  }
#pragma unroll
  for (int m = 1; m < 16; m <<= 1)
#pragma unroll
    for (int r = 0; r < 8; ++r) pool[r] = fmaxf(pool[r], __shfl_xor(pool[r], m, 32));
  if (nn == 0) {
    size_t base = (((size_t)pz * 3 + py) * 3 + px) * 64;
#pragma unroll
    for (int r = 0; r < 8; ++r) f3[base + mt * 16 + r + 8 * hf] = pool[r];
  }
}

// ---------------- kernel 4: gather + trilinear sample + MLP -------------------------
__global__ void __launch_bounds__(128) k_decode(const float* __restrict__ qp,
                                                const int* __restrict__ sel,
                                                const float* __restrict__ f3,
                                                const __bf16* __restrict__ w1,
                                                const float* __restrict__ b1,
                                                const __bf16* __restrict__ w2,
                                                const float* __restrict__ b2,
                                                const float* __restrict__ w3,
                                                const float* __restrict__ b3,
                                                float* __restrict__ out) {
  __shared__ __align__(16) unsigned hbuf[4][16][16];
  int lane = threadIdx.x & 31, wave = threadIdx.x >> 5;
  int tile = blockIdx.x * 4 + wave;  // 16 queries per wave-tile
  if (tile >= 625) return;           // wave-uniform exit: EXEC stays all-ones
  int hf = lane >> 4, nn = lane & 15;
  int q = tile * 16 + nn;
  int sid = sel[q];
  float qx = qp[sid * 3 + 0], qy = qp[sid * 3 + 1], qz = qp[sid * 3 + 2];
  float ix = qx + 1.f, iy = qy + 1.f, iz = qz + 1.f;  // (q+1)*0.5*(3-1)
  float fx = floorf(ix), fy = floorf(iy), fz = floorf(iz);
  float wx = ix - fx, wy = iy - fy, wz = iz - fz;
  int x0i = iclamp((int)fx, 0, 2), y0i = iclamp((int)fy, 0, 2), z0i = iclamp((int)fz, 0, 2);
  int x1i = iclamp(x0i + 1, 0, 2), y1i = iclamp(y0i + 1, 0, 2), z1i = iclamp(z0i + 1, 0, 2);
  int xs[2] = {x0i, x1i}, ys[2] = {y0i, y1i}, zs[2] = {z0i, z1i};
  float wxs[2] = {1.f - wx, wx}, wys[2] = {1.f - wy, wy}, wzs[2] = {1.f - wz, wz};
  int cb[8];
  float cw[8];
#pragma unroll
  for (int zj = 0; zj < 2; ++zj)
#pragma unroll
    for (int yj = 0; yj < 2; ++yj)
#pragma unroll
      for (int xj = 0; xj < 2; ++xj) {
        int j = zj * 4 + yj * 2 + xj;
        cb[j] = ((zs[zj] * 3 + ys[yj]) * 3 + xs[xj]) * 64;
        cw[j] = wzs[zj] * wys[yj] * wxs[xj];
      }

  // dec1: 32x64 (2 M-tiles, 2 K-steps), B built directly from trilinear samples
  const unsigned* w1u = (const unsigned*)w1;
  v8f acc[2];
#pragma unroll
  for (int mt = 0; mt < 2; ++mt)
#pragma unroll
    for (int r = 0; r < 8; ++r) acc[mt][r] = b1[mt * 16 + r + 8 * hf];
#pragma unroll
  for (int s = 0; s < 2; ++s) {
    BFrag B;
#pragma unroll
    for (int v = 0; v < 8; ++v) {
      int c0 = 32 * s + 16 * hf + 2 * v;
      float v0 = 0.f, v1 = 0.f;
#pragma unroll
      for (int j = 0; j < 8; ++j) {
        const float2 f = *(const float2*)(f3 + cb[j] + c0);
        v0 += cw[j] * f.x;
        v1 += cw[j] * f.y;
      }
      B.u[v] = pack2((__bf16)v0, (__bf16)v1);
    }
#pragma unroll
    for (int mt = 0; mt < 2; ++mt) {
      BFrag A;
#pragma unroll
      for (int v = 0; v < 8; ++v) {
        int K = 32 * s + 16 * hf + 2 * v;
        A.u[v] = w1u[((mt * 16 + nn) * 64 + K) >> 1];
      }
      acc[mt] = __builtin_amdgcn_wmma_f32_16x16x32_bf16(false, A.v, false, B.v,
                                                        (short)0, acc[mt], false, false);
    }
  }
  // relu -> h1 in LDS (bf16 pairs, [q][ch])
#pragma unroll
  for (int mt = 0; mt < 2; ++mt)
#pragma unroll
    for (int r = 0; r < 8; r += 2) {
      float a0 = fmaxf(acc[mt][r], 0.f), a1 = fmaxf(acc[mt][r + 1], 0.f);
      hbuf[wave][nn][mt * 8 + 4 * hf + (r >> 1)] = pack2((__bf16)a0, (__bf16)a1);
    }
  asm volatile("s_wait_dscnt 0x0" ::: "memory");

  // dec2: 16x32, single WMMA
  const unsigned* w2u = (const unsigned*)w2;
  v8f acc2;
#pragma unroll
  for (int r = 0; r < 8; ++r) acc2[r] = b2[r + 8 * hf];
  {
    BFrag A, B;
#pragma unroll
    for (int v = 0; v < 8; ++v) {
      int K = 2 * v + 16 * hf;
      A.u[v] = w2u[(nn * 32 + K) >> 1];
      B.u[v] = hbuf[wave][nn][8 * hf + v];  // K = 16*hf + 2v
    }
    acc2 = __builtin_amdgcn_wmma_f32_16x16x32_bf16(false, A.v, false, B.v,
                                                   (short)0, acc2, false, false);
  }
#pragma unroll
  for (int r = 0; r < 8; r += 2) {
    float a0 = fmaxf(acc2[r], 0.f), a1 = fmaxf(acc2[r + 1], 0.f);
    hbuf[wave][nn][4 * hf + (r >> 1)] = pack2((__bf16)a0, (__bf16)a1);
  }
  asm volatile("s_wait_dscnt 0x0" ::: "memory");

  // dec3: 2x16, scalar per lane (o = hf, query = nn)
  float s3 = b3[hf];
#pragma unroll
  for (int k = 0; k < 16; k += 2) {
    unsigned u = hbuf[wave][nn][k >> 1];
    s3 += w3[hf * 16 + k] * bflo(u) + w3[hf * 16 + k + 1] * bfhi(u);
  }
  out[hf * 10000 + tile * 16 + nn] = s3;
}

// ---------------- launch ------------------------------------------------------------
extern "C" void kernel_launch(void* const* d_in, const int* in_sizes, int n_in,
                              void* d_out, int out_size, void* d_ws, size_t ws_size,
                              hipStream_t stream) {
  const float* x0  = (const float*)d_in[0];
  const float* x1  = (const float*)d_in[1];
  const float* qp  = (const float*)d_in[2];
  const int*   sel = (const int*)d_in[3];
  const float* e1w = (const float*)d_in[4];
  const float* e1b = (const float*)d_in[5];
  const float* e2w = (const float*)d_in[6];
  const float* e2b = (const float*)d_in[7];
  const float* e3w = (const float*)d_in[8];
  const float* e3b = (const float*)d_in[9];
  const float* d1w = (const float*)d_in[10];
  const float* d1b = (const float*)d_in[11];
  const float* d2w = (const float*)d_in[12];
  const float* d2b = (const float*)d_in[13];
  const float* d3w = (const float*)d_in[14];
  const float* d3b = (const float*)d_in[15];

  char* ws = (char*)d_ws;
  __bf16* e1p = (__bf16*)(ws + OFF_E1P);
  __bf16* e2p = (__bf16*)(ws + OFF_E2P);
  __bf16* e3p = (__bf16*)(ws + OFF_E3P);
  __bf16* d1p = (__bf16*)(ws + OFF_D1P);
  __bf16* d2p = (__bf16*)(ws + OFF_D2P);
  float*  f3  = (float*)(ws + OFF_F3);
  __bf16* f2  = (__bf16*)(ws + OFF_F2);
  __bf16* f1  = (__bf16*)(ws + OFF_F1);
  float*  out = (float*)d_out;

  k_pack<<<284, 256, 0, stream>>>(e1w, e2w, e3w, d1w, d2w, e1p, e2p, e3p, d1p, d2p);
  k_conv1<<<1728, 256, 0, stream>>>(x0, x1, e1p, e1b, f1);
  k_conv2<<<216, 256, 0, stream>>>(f1, e2p, e2b, f2);
  k_conv3<<<27, 128, 0, stream>>>(f2, e3p, e3b, f3);
  k_decode<<<157, 128, 0, stream>>>(qp, sel, f3, d1p, d1b, d2p, d2b, d3w, d3b, out);
}
